// GeometryAnomalyDetector_73383811219658
// MI455X (gfx1250) — compile-verified, matrix operand order checked by reading, not run on hardware
//
#include <hip/hip_runtime.h>
#include <hip/hip_bf16.h>

typedef _Float16 f16;
typedef __attribute__((ext_vector_type(16))) _Float16 v16h;
typedef __attribute__((ext_vector_type(8)))  _Float16 v8h;
typedef __attribute__((ext_vector_type(8)))  float    v8f;

#define MPTS   32768      // B*N
#define BEPS   1e-5f

// ---------------------------------------------------------------------------
// Pack points (M x 3 fp32) into f16 activation buffer padded to Kpad=32
// ---------------------------------------------------------------------------
__global__ __launch_bounds__(256) void pack_points(const float* __restrict__ pts,
                                                   f16* __restrict__ X) {
    int idx = blockIdx.x * 256 + threadIdx.x;       // M*32 elements
    int m = idx >> 5, j = idx & 31;
    float v = (j < 3) ? pts[(size_t)m * 3 + j] : 0.f;
    X[idx] = (f16)v;
}

// ---------------------------------------------------------------------------
// Convert/pad/transpose weights: W[Cin x Cout] fp32 -> Wt[Cout x Kpad] f16
// ---------------------------------------------------------------------------
__global__ __launch_bounds__(256) void cvt_weight(const float* __restrict__ W,
                                                  f16* __restrict__ Wt,
                                                  int Cin, int Cout, int Kpad) {
    int idx = blockIdx.x * 256 + threadIdx.x;       // Cout*Kpad elements
    if (idx >= Cout * Kpad) return;
    int co = idx / Kpad, ci = idx % Kpad;
    float v = (ci < Cin) ? W[(size_t)ci * Cout + co] : 0.f;
    Wt[idx] = (f16)v;
}

__global__ __launch_bounds__(256) void zero_u32(unsigned* __restrict__ p, int n) {
    int i = blockIdx.x * 256 + threadIdx.x;
    if (i < n) p[i] = 0u;
}

// ---------------------------------------------------------------------------
// WMMA GEMM, register-blocked 64x32 per wave (4 M-frags x 2 N-frags).
// Templated on (KPAD, COUT): all fragment strides are compile-time immediates
// -> no per-iteration 64-bit VALU address math, no WMMA WAR-hazard NOPs.
// Per K=32 step: 12 b128 loads feed 8 v_wmma ops.
// ---------------------------------------------------------------------------
template<int KPAD, int COUT>
__global__ __launch_bounds__(128) void wmma_gemm_bias(const f16* __restrict__ X,
                                                      const f16* __restrict__ Wt,
                                                      const float* __restrict__ bias,
                                                      float* __restrict__ Y) {
    const int lane  = threadIdx.x & 31;
    const int wave  = threadIdx.x >> 5;
    const int tileN = blockIdx.x;                 // 32-column strip
    const int strip = blockIdx.y * 4 + wave;      // 64-row strip
    const int l15   = lane & 15;
    const int hi16  = lane >> 4;                  // 0 | 1
    const int kbase = hi16 * 8;                   // ISA 16-bit fragment layout
    constexpr size_t FRAG = (size_t)16 * KPAD;    // halfs between fragments

    const f16* xrow = X  + (size_t)(strip * 64 + l15) * KPAD + kbase;
    const f16* wrow = Wt + (size_t)(tileN * 32 + l15) * KPAD + kbase;

    const float bn0 = bias[tileN * 32 + l15];
    const float bn1 = bias[tileN * 32 + 16 + l15];
    v8f acc[4][2];
#pragma unroll
    for (int i = 0; i < 4; ++i)
#pragma unroll
        for (int r = 0; r < 8; ++r) { acc[i][0][r] = bn0; acc[i][1][r] = bn1; }

#pragma unroll 2
    for (int k0 = 0; k0 < KPAD; k0 += 32) {
        v16h a[4], b[2];
#pragma unroll
        for (int i = 0; i < 4; ++i) {
            v8h lo = *(const v8h*)(xrow + i * FRAG + k0);
            v8h hi = *(const v8h*)(xrow + i * FRAG + k0 + 16);
#pragma unroll
            for (int t = 0; t < 8; ++t) { a[i][t] = lo[t]; a[i][8 + t] = hi[t]; }
        }
#pragma unroll
        for (int j = 0; j < 2; ++j) {
            v8h lo = *(const v8h*)(wrow + j * FRAG + k0);
            v8h hi = *(const v8h*)(wrow + j * FRAG + k0 + 16);
#pragma unroll
            for (int t = 0; t < 8; ++t) { b[j][t] = lo[t]; b[j][8 + t] = hi[t]; }
        }
#pragma unroll
        for (int i = 0; i < 4; ++i)
#pragma unroll
            for (int j = 0; j < 2; ++j)
                acc[i][j] = __builtin_amdgcn_wmma_f32_16x16x32_f16(
                    false, a[i], false, b[j], (short)0, acc[i][j], false, false);
    }

    // D layout: lane<16 -> rows 0..7, lane>=16 -> rows 8..15 ; col = lane&15
    float* ybase = Y + (size_t)(strip * 64 + hi16 * 8) * COUT + tileN * 32 + l15;
#pragma unroll
    for (int i = 0; i < 4; ++i)
#pragma unroll
        for (int j = 0; j < 2; ++j)
#pragma unroll
            for (int r = 0; r < 8; ++r)
                ybase[(size_t)(i * 16 + r) * COUT + j * 16] = acc[i][j][r];
}

// ---------------------------------------------------------------------------
// Per-channel partial sum/sumsq over a 128-row strip (coalesced over channels)
// ---------------------------------------------------------------------------
__global__ __launch_bounds__(256) void col_partial(const float* __restrict__ Y,
                                                   int Cout,
                                                   float* __restrict__ sum,
                                                   float* __restrict__ sumsq) {
    const int row0 = blockIdx.x * 128;
    for (int c = threadIdx.x; c < Cout; c += 256) {
        float s = 0.f, s2 = 0.f;
        const float* yp = Y + (size_t)row0 * Cout + c;
#pragma unroll 4
        for (int r = 0; r < 128; ++r) {
            float v = yp[(size_t)r * Cout];
            s += v; s2 += v * v;
        }
        atomicAdd(&sum[c], s);
        atomicAdd(&sumsq[c], s2);
    }
}

// ---------------------------------------------------------------------------
// BN + ReLU + cast to f16 for the next layer; optional max-pool into featbits
// ---------------------------------------------------------------------------
template<int CSHIFT, bool FEAT>
__global__ __launch_bounds__(256) void bn_relu_cvt(const float* __restrict__ Y,
                                                   const float* __restrict__ sum,
                                                   const float* __restrict__ sumsq,
                                                   const float* __restrict__ g,
                                                   const float* __restrict__ be,
                                                   f16* __restrict__ Xn,
                                                   unsigned* __restrict__ featbits) {
    const int idx = blockIdx.x * 256 + threadIdx.x;   // M*Cout
    const int c = idx & ((1 << CSHIFT) - 1);
    constexpr float invM = 1.f / (float)MPTS;
    float m  = sum[c] * invM;
    float vr = sumsq[c] * invM - m * m;
    float scale = g[c] * rsqrtf(vr + BEPS);
    float shift = be[c] - m * scale;
    float r = fmaxf(Y[idx] * scale + shift, 0.f);
    Xn[idx] = (f16)r;
    if (FEAT) {
        int b = idx >> (CSHIFT + 11);                 // point index / 2048
        atomicMax(&featbits[((size_t)b << CSHIFT) + c], __float_as_uint(r)); // r >= 0
    }
}

// ---------------------------------------------------------------------------
// Tiny FC stack (single workgroup): feat->256->z(64)->256->512
// ---------------------------------------------------------------------------
__global__ __launch_bounds__(512) void fc_stack(const unsigned* __restrict__ featbits,
                                                const float* __restrict__ W0, const float* __restrict__ b0,
                                                const float* __restrict__ W1, const float* __restrict__ b1,
                                                const float* __restrict__ W2, const float* __restrict__ b2,
                                                const float* __restrict__ W3, const float* __restrict__ b3,
                                                float* __restrict__ z_out,
                                                float* __restrict__ hdec) {
    __shared__ float bufA[16 * 512];   // feat, then z
    __shared__ float bufB[16 * 256];   // h1, then h2
    const int tid = threadIdx.x;

    for (int i = tid; i < 16 * 512; i += 512) bufA[i] = __uint_as_float(featbits[i]);
    __syncthreads();
    for (int i = tid; i < 16 * 256; i += 512) {        // h1 = relu(feat@W0+b0)
        int r = i >> 8, c = i & 255;
        float s = b0[c];
        for (int k = 0; k < 512; ++k) s += bufA[r * 512 + k] * W0[(size_t)k * 256 + c];
        bufB[i] = fmaxf(s, 0.f);
    }
    __syncthreads();
    for (int i = tid; i < 16 * 64; i += 512) {         // z = h1@W1+b1
        int r = i >> 6, c = i & 63;
        float s = b1[c];
        for (int k = 0; k < 256; ++k) s += bufB[r * 256 + k] * W1[(size_t)k * 64 + c];
        z_out[i] = s;
        bufA[i]  = s;
    }
    __syncthreads();
    for (int i = tid; i < 16 * 256; i += 512) {        // h2 = relu(z@W2+b2)
        int r = i >> 8, c = i & 255;
        float s = b2[c];
        for (int k = 0; k < 64; ++k) s += bufA[r * 64 + k] * W2[(size_t)k * 256 + c];
        bufB[i] = fmaxf(s, 0.f);
    }
    __syncthreads();
    for (int i = tid; i < 16 * 512; i += 512) {        // h3 = relu(h2@W3+b3)
        int r = i >> 9, c = i & 511;
        float s = b3[c];
        for (int k = 0; k < 256; ++k) s += bufB[r * 256 + k] * W3[(size_t)k * 512 + c];
        hdec[i] = fmaxf(s, 0.f);
    }
}

// ---------------------------------------------------------------------------
// Broadcast hdec[b][c] across the N dimension into f16 decoder input
// ---------------------------------------------------------------------------
__global__ __launch_bounds__(256) void broadcast_h(const float* __restrict__ hdec,
                                                   f16* __restrict__ X) {
    int idx = blockIdx.x * 256 + threadIdx.x;      // M*512
    int m = idx >> 9, c = idx & 511;
    X[idx] = (f16)hdec[((size_t)(m >> 11) << 9) + c];
}

// ---------------------------------------------------------------------------
// Final 128 -> 3 linear layer, writes reconstructed points
// ---------------------------------------------------------------------------
__global__ __launch_bounds__(256) void final_linear3(const f16* __restrict__ X,
                                                     const float* __restrict__ W,
                                                     const float* __restrict__ bias,
                                                     float* __restrict__ recon) {
    int m = blockIdx.x * 256 + threadIdx.x;
    if (m >= MPTS) return;
    const f16* xr = X + (size_t)m * 128;
    float s0 = bias[0], s1 = bias[1], s2 = bias[2];
#pragma unroll 4
    for (int k = 0; k < 128; ++k) {
        float x = (float)xr[k];
        s0 += x * W[k * 3 + 0];
        s1 += x * W[k * 3 + 1];
        s2 += x * W[k * 3 + 2];
    }
    recon[(size_t)m * 3 + 0] = s0;
    recon[(size_t)m * 3 + 1] = s1;
    recon[(size_t)m * 3 + 2] = s2;
}

// ---------------------------------------------------------------------------
// Chamfer: for each row i of A, min_j |A_i - B_j|^2 (run twice, roles swapped)
// ---------------------------------------------------------------------------
__global__ __launch_bounds__(256) void nn_rowmin(const float* __restrict__ A,
                                                 const float* __restrict__ Bp,
                                                 float* __restrict__ outmin) {
    const int b = blockIdx.y;
    const int n = blockIdx.x * 256 + threadIdx.x;
    const float* ap = A + ((size_t)b * 2048 + n) * 3;
    float px = ap[0], py = ap[1], pz = ap[2];
    __shared__ float tx[256], ty[256], tz[256];
    float best = 3.4e38f;
    for (int mt = 0; mt < 2048; mt += 256) {
        const float* bp = Bp + ((size_t)b * 2048 + mt + threadIdx.x) * 3;
        tx[threadIdx.x] = bp[0]; ty[threadIdx.x] = bp[1]; tz[threadIdx.x] = bp[2];
        __syncthreads();
#pragma unroll 4
        for (int j = 0; j < 256; ++j) {
            float dx = px - tx[j], dy = py - ty[j], dz = pz - tz[j];
            best = fminf(best, dx * dx + dy * dy + dz * dz);
        }
        __syncthreads();
    }
    outmin[(size_t)b * 2048 + n] = best;
}

__global__ __launch_bounds__(256) void loss_reduce(const float* __restrict__ minn,
                                                   const float* __restrict__ minm,
                                                   float* __restrict__ out) {
    float s = 0.f;
    for (int i = threadIdx.x; i < 16 * 2048; i += 256) s += minn[i] + minm[i];
    __shared__ float ss[256];
    ss[threadIdx.x] = s;
    __syncthreads();
    for (int st = 128; st > 0; st >>= 1) {
        if (threadIdx.x < st) ss[threadIdx.x] += ss[threadIdx.x + st];
        __syncthreads();
    }
    if (threadIdx.x == 0) out[0] = ss[0] / (16.f * 2048.f);
}

// ---------------------------------------------------------------------------
// Host orchestration
// ---------------------------------------------------------------------------
extern "C" void kernel_launch(void* const* d_in, const int* in_sizes, int n_in,
                              void* d_out, int out_size, void* d_ws, size_t ws_size,
                              hipStream_t stream) {
    const float* pts = (const float*)d_in[0];
    // params: enc i -> W=1+4i,b,g,be ; fce: 17..20 ; fcd: 21..24 ; dec: 25..34
    auto F = [&](int i) { return (const float*)d_in[i]; };

    char* ws = (char*)d_ws;
    size_t o = 0;
    auto take = [&](size_t bytes) { char* p = ws + o; o += (bytes + 255) & ~(size_t)255; return p; };

    f16*   Xa    = (f16*)  take((size_t)MPTS * 512 * 2);
    f16*   Xb    = (f16*)  take((size_t)MPTS * 512 * 2);
    float* Y     = (float*)take((size_t)MPTS * 512 * 4);
    float* sum   = (float*)take(512 * 4);
    float* sumsq = (float*)take(512 * 4);
    unsigned* featbits = (unsigned*)take(16 * 512 * 4);
    float* hdec  = (float*)take(16 * 512 * 4);
    float* minn  = (float*)take(16 * 2048 * 4);
    float* minm  = (float*)take(16 * 2048 * 4);
    f16* we0 = (f16*)take((size_t)64  * 32  * 2);
    f16* we1 = (f16*)take((size_t)128 * 64  * 2);
    f16* we2 = (f16*)take((size_t)256 * 128 * 2);
    f16* we3 = (f16*)take((size_t)512 * 256 * 2);
    f16* wd0 = (f16*)take((size_t)256 * 512 * 2);
    f16* wd1 = (f16*)take((size_t)128 * 256 * 2);

    float* out   = (float*)d_out;
    float* recon = out + 1;              // 16*2048*3
    float* z_out = out + 1 + MPTS * 3;   // 16*64

    auto cvt = [&](const float* W, f16* Wt, int Cin, int Cout, int Kpad) {
        int n = Cout * Kpad;
        cvt_weight<<<(n + 255) / 256, 256, 0, stream>>>(W, Wt, Cin, Cout, Kpad);
    };
    cvt(F(1),  we0, 3,   64,  32);
    cvt(F(5),  we1, 64,  128, 64);
    cvt(F(9),  we2, 128, 256, 128);
    cvt(F(13), we3, 256, 512, 256);
    cvt(F(25), wd0, 512, 256, 512);
    cvt(F(29), wd1, 256, 128, 256);

    zero_u32<<<(16 * 512 + 255) / 256, 256, 0, stream>>>(featbits, 16 * 512);
    pack_points<<<(MPTS * 32) / 256, 256, 0, stream>>>(pts, Xa);

    // --- one MLP layer: GEMM + stats + BN/ReLU/cast (compile-time shapes) ---
    #define RUN_LAYER(KPAD, COUT, CSHIFT, Xin, Wt, bW, Xnext, FEAT, fb)            \
        do {                                                                       \
            zero_u32<<<4, 256, 0, stream>>>((unsigned*)sum, 2 * 512);              \
            dim3 g((COUT) / 32, MPTS / 256);                                       \
            wmma_gemm_bias<KPAD, COUT><<<g, 128, 0, stream>>>(Xin, Wt,             \
                                                              F(bW + 1), Y);       \
            col_partial<<<MPTS / 128, 256, 0, stream>>>(Y, COUT, sum, sumsq);      \
            bn_relu_cvt<CSHIFT, FEAT><<<(MPTS * (COUT)) / 256, 256, 0, stream>>>(  \
                Y, sum, sumsq, F(bW + 2), F(bW + 3), Xnext, fb);                   \
        } while (0)

    // encoder (3->64->128->256->512); enc3 feeds the max-pool
    RUN_LAYER(32,  64,  6, Xa, we0, 1,  Xb, false, nullptr);
    RUN_LAYER(64,  128, 7, Xb, we1, 5,  Xa, false, nullptr);
    RUN_LAYER(128, 256, 8, Xa, we2, 9,  Xb, false, nullptr);
    RUN_LAYER(256, 512, 9, Xb, we3, 13, Xa, true,  featbits);

    // FC stack: feat -> 256 -> z(64) -> 256 -> 512
    fc_stack<<<1, 512, 0, stream>>>(featbits,
                                    F(17), F(18), F(19), F(20),
                                    F(21), F(22), F(23), F(24),
                                    z_out, hdec);

    broadcast_h<<<(MPTS * 512) / 256, 256, 0, stream>>>(hdec, Xb);

    // decoder (512->256->128), then 128->3
    RUN_LAYER(512, 256, 8, Xb, wd0, 25, Xa, false, nullptr);
    RUN_LAYER(256, 128, 7, Xa, wd1, 29, Xb, false, nullptr);
    final_linear3<<<MPTS / 256, 256, 0, stream>>>(Xb, F(33), F(34), recon);

    // chamfer: two role-swapped row-min passes + reduction
    dim3 cg(2048 / 256, 16);
    nn_rowmin<<<cg, 256, 0, stream>>>(recon, pts, minn);
    nn_rowmin<<<cg, 256, 0, stream>>>(pts, recon, minm);
    loss_reduce<<<1, 256, 0, stream>>>(minn, minm, out);

    #undef RUN_LAYER
}
